// MultiHeadCDGCN_18030272709046
// MI455X (gfx1250) — compile-verified, matrix operand order checked by reading, not run on hardware
//
#include <hip/hip_runtime.h>
#include <hip/hip_bf16.h>
#include <math.h>

// ---- problem constants (reference: B=8, T=32, N=256, D=256, H=8, dh=32) ----
#define BB 8
#define TT 32
#define NN 256
#define DD 256
#define HH 8
#define DH 32

typedef __attribute__((ext_vector_type(16))) _Float16 v16h;
typedef __attribute__((ext_vector_type(8)))  float    v8f;

union FragAB { v16h h; unsigned u[8]; };

// Starting k of the consecutive-k pair held in VGPR v for this lane-half,
// per CDNA5 ISA 16-bit A/B 16x32 layout:
//  lanes 0-15:  v0..v3 -> k=0,2,4,6 ; v4..v7 -> k=16,18,20,22
//  lanes 16-31: v0..v3 -> k=8..14   ; v4..v7 -> k=24..30
__device__ __forceinline__ int kbase(int v, int half) {
    return ((v & 4) << 2) + (half << 3) + ((v & 3) << 1);
}

// ---------------------------------------------------------------------------
// Kernel 1: convert the three 256x256 f32 weight matrices to f16
// ---------------------------------------------------------------------------
__global__ __launch_bounds__(256)
void cvt_w_kernel(const float* __restrict__ wq, const float* __restrict__ wk,
                  const float* __restrict__ wv,
                  _Float16* __restrict__ q, _Float16* __restrict__ k,
                  _Float16* __restrict__ v) {
    int i = blockIdx.x * 256 + threadIdx.x;   // grid covers 65536 exactly
    q[i] = (_Float16)wq[i];
    k[i] = (_Float16)wk[i];
    v[i] = (_Float16)wv[i];
}

// ---------------------------------------------------------------------------
// Kernel 2: temporal softmax pooling over T, fused with x -> f16 conversion.
// TAtt[b,n,d] = sum_t x[b,t,n,d] * softmax_t(x[b,:,n,d])  (stored f16)
// Xh[b,t,n,d] = (f16) x[b,t,n,d]   (free on the write side: x already read)
// ---------------------------------------------------------------------------
__global__ __launch_bounds__(256)
void temporal_pool_kernel(const float* __restrict__ x,
                          _Float16* __restrict__ Xh,
                          _Float16* __restrict__ TAtt) {
    int g = blockIdx.x * 256 + threadIdx.x;          // < B*N*D = 524288
    int d = g & 255;
    int n = (g >> 8) & 255;
    int b = g >> 16;
    const size_t base = (size_t)b * (TT * NN * DD) + (size_t)n * DD + d;
    float vals[TT];
    float mx = -3.4e38f;
#pragma unroll
    for (int t = 0; t < TT; ++t) {
        vals[t] = x[base + (size_t)t * (NN * DD)];
        mx = fmaxf(mx, vals[t]);
    }
    float Z = 0.f, S = 0.f;
#pragma unroll
    for (int t = 0; t < TT; ++t) {
        float e = expf(vals[t] - mx);
        Z += e;
        S += e * vals[t];
        Xh[base + (size_t)t * (NN * DD)] = (_Float16)vals[t];
    }
    TAtt[g] = (_Float16)(S / Z);
}

// ---------------------------------------------------------------------------
// Kernel 3: C_f16[M,256] = A_f16[M,256] @ W_f16^T   (y = x @ W.T, no bias)
// Block = 128 threads (4 waves); block tile 16 rows x 256 cols (64/wave).
// A row-tile staged to LDS once via GLOBAL_LOAD_ASYNC_TO_LDS_B128 (ASYNCcnt).
// mode 0: C row-major [row][j].  mode 1 (for V): transposed [b][j][n].
// ---------------------------------------------------------------------------
#define LDSK 264   // 256 + 8 halfs pad: row stride 528 B (16B-aligned,
                   // conflict-free for the strided fragment reads)

__global__ __launch_bounds__(128)
void gemm_xwT_kernel(const _Float16* __restrict__ A, const _Float16* __restrict__ Wh,
                     _Float16* __restrict__ C, int mode) {
    __shared__ _Float16 sA[16 * LDSK];
    const int tid  = threadIdx.x;
    const int lane = tid & 31;
    const int wave = tid >> 5;
    const int half = lane >> 4;
    const int l16  = lane & 15;
    const int m0   = blockIdx.x * 16;

    // async byte-copy of the 16x256 f16 A tile into padded LDS rows:
    // 512 chunks of 16B; 4 per thread. Both sides 16B aligned.
    {
        const unsigned base_lds = (unsigned)(size_t)&sA[0];
#pragma unroll
        for (int i = 0; i < 4; ++i) {
            const int q = tid + 128 * i;
            const int r = q >> 5;            // row 0..15
            const int c = (q & 31) * 8;      // col in halfs, 16B granules
            unsigned lds_addr = base_lds + (unsigned)(r * LDSK + c) * 2u;
            unsigned long long gaddr =
                (unsigned long long)(const void*)&A[(size_t)(m0 + r) * DD + c];
            asm volatile("global_load_async_to_lds_b128 %0, %1, off"
                         :: "v"(lds_addr), "v"(gaddr) : "memory");
        }
        asm volatile("s_wait_asynccnt 0x0" ::: "memory");
    }

    // columns handled by this wave: j(cc) = wave*64 + cc*16 + l16
    const _Float16* wrow[4];
#pragma unroll
    for (int cc = 0; cc < 4; ++cc) {
        wrow[cc] = Wh + (size_t)(wave * 64 + cc * 16 + l16) * DD;  // B[k][j]=W[j][k]
        __builtin_prefetch(wrow[cc], 0, 3);   // global_prefetch_b8 (L2 warm)
    }
    __syncthreads();

    v8f acc[4] = {{}, {}, {}, {}};
#pragma unroll 2
    for (int ks = 0; ks < 8; ++ks) {
        FragAB a, b[4];
#pragma unroll
        for (int v = 0; v < 8; ++v)
            a.u[v] = *(const unsigned*)&sA[l16 * LDSK + ks * 32 + kbase(v, half)];
#pragma unroll
        for (int cc = 0; cc < 4; ++cc)
#pragma unroll
            for (int v = 0; v < 8; ++v)
                b[cc].u[v] = *(const unsigned*)&wrow[cc][ks * 32 + kbase(v, half)];
#pragma unroll
        for (int cc = 0; cc < 4; ++cc)
            acc[cc] = __builtin_amdgcn_wmma_f32_16x16x32_f16(
                false, a.h, false, b[cc].h, (short)0, acc[cc], false, false);
    }

#pragma unroll
    for (int cc = 0; cc < 4; ++cc) {
        const int j = wave * 64 + cc * 16 + l16;
#pragma unroll
        for (int r = 0; r < 8; ++r) {
            const int row = m0 + r + 8 * half;
            if (mode == 0) {
                C[(size_t)row * DD + j] = (_Float16)acc[cc][r];
            } else {
                // row = b*256 + n  ->  C[b][j][n]
                const int bb_ = row >> 8, nn_ = row & 255;
                C[((size_t)bb_ * DD + j) * NN + nn_] = (_Float16)acc[cc][r];
            }
        }
    }
}

// ---------------------------------------------------------------------------
// Kernel 4: attention. One wave per (b, t, h, n-tile of 16).
// Phase 1: 16 score WMMAs: A = relu(Q K^T / sqrt(dh)) + I, written f16 to LDS.
// Phase 2: out = A @ V with V pre-transposed (Vt[b][h*32+d][m]) so B-fragments
//          are packed k-pair loads (b128-mergeable). 8 K-steps x 2 d-tiles.
// ---------------------------------------------------------------------------
#define SSTR 264   // padded score row stride (halfs)

__global__ __launch_bounds__(128)
void attn_kernel(const _Float16* __restrict__ Qh, const _Float16* __restrict__ Kh,
                 const _Float16* __restrict__ Vt, float* __restrict__ out) {
    __shared__ _Float16 sS[4][16 * SSTR];
    const int tid  = threadIdx.x;
    const int lane = tid & 31;
    const int wave = tid >> 5;
    const int half = lane >> 4;
    const int l16  = lane & 15;

    const unsigned unit = blockIdx.x * 4 + wave;   // < B*T*H*(N/16) = 32768
    const int nt = unit & 15;
    const int h  = (unit >> 4) & 7;
    const int t  = (unit >> 7) & 31;
    const int b  = unit >> 12;
    const int n0 = nt * 16;
    _Float16* S = sS[wave];

    // Q fragment: rows n0+l16, k = d (dh=32 -> single WMMA K-step); reused 16x
    FragAB aQ;
    {
        const _Float16* qrow =
            Qh + ((size_t)((b * TT + t) * NN + n0 + l16)) * DD + h * DH;
#pragma unroll
        for (int v = 0; v < 8; ++v) aQ.u[v] = *(const unsigned*)&qrow[kbase(v, half)];
    }
    const float scale = 0.17677669529663687f;   // 1/sqrt(32)

    // ---- Phase 1: scores ----
#pragma unroll 2
    for (int mt = 0; mt < 16; ++mt) {
        FragAB bK;  // B[k=d][col=m] = K[m][d] : per-lane contiguous in d
        const _Float16* krow =
            Kh + ((size_t)(b * NN + mt * 16 + l16)) * DD + h * DH;
#pragma unroll
        for (int v = 0; v < 8; ++v) bK.u[v] = *(const unsigned*)&krow[kbase(v, half)];
        v8f acc = {};
        acc = __builtin_amdgcn_wmma_f32_16x16x32_f16(false, aQ.h, false, bK.h,
                                                     (short)0, acc, false, false);
        const int m = mt * 16 + l16;            // D-frag: lane = col m
#pragma unroll
        for (int r = 0; r < 8; ++r) {           // D-frag: vgpr r = row n
            int n = n0 + r + 8 * half;
            float sv = fmaxf(acc[r] * scale, 0.f) + ((n == m) ? 1.f : 0.f);
            S[(r + 8 * half) * SSTR + m] = (_Float16)sv;
        }
    }
    __syncthreads();   // LDS visibility before transposed fragment reads

    // ---- Phase 2: out = A @ V ----
    const _Float16* vt0 = Vt + ((size_t)(b * DD) + h * DH + l16) * NN;      // d tile 0
    const _Float16* vt1 = vt0 + 16 * NN;                                    // d tile 1
    v8f o0 = {}, o1 = {};
#pragma unroll 2
    for (int ms = 0; ms < 8; ++ms) {
        FragAB aS;  // rows = n (lane l16), k = m (packed pairs from LDS)
        FragAB bV0, bV1;  // B[k=m][col=d] = Vt[d][m] : per-lane contiguous in m
#pragma unroll
        for (int v = 0; v < 8; ++v) {
            const int k = ms * 32 + kbase(v, half);
            aS.u[v]  = *(const unsigned*)&S[l16 * SSTR + k];
            bV0.u[v] = *(const unsigned*)&vt0[k];
            bV1.u[v] = *(const unsigned*)&vt1[k];
        }
        o0 = __builtin_amdgcn_wmma_f32_16x16x32_f16(false, aS.h, false, bV0.h,
                                                    (short)0, o0, false, false);
        o1 = __builtin_amdgcn_wmma_f32_16x16x32_f16(false, aS.h, false, bV1.h,
                                                    (short)0, o1, false, false);
    }
    float* obase = out + ((size_t)((b * TT + t) * NN)) * DD + h * DH;
#pragma unroll
    for (int r = 0; r < 8; ++r) {
        int n = n0 + r + 8 * half;
        obase[(size_t)n * DD + l16]      = o0[r];
        obase[(size_t)n * DD + 16 + l16] = o1[r];
    }
}

// ---------------------------------------------------------------------------
extern "C" void kernel_launch(void* const* d_in, const int* in_sizes, int n_in,
                              void* d_out, int out_size, void* d_ws, size_t ws_size,
                              hipStream_t stream) {
    const float* x  = (const float*)d_in[0];
    // d_in[1] = boxes_in_flat : unused by the reference forward
    const float* WQ = (const float*)d_in[2];
    const float* WK = (const float*)d_in[3];
    const float* WV = (const float*)d_in[4];
    float* out = (float*)d_out;

    // workspace layout (~71 MB)
    char* ws = (char*)d_ws;
    _Float16* Xh   = (_Float16*)ws;                           // 65536*256 f16 = 33.5 MB
    _Float16* TAtt = Xh + (size_t)65536 * 256;                // 2048*256 f16 = 1 MB
    _Float16* Wqh  = TAtt + 2048 * 256;                       // 3 x 128 KB
    _Float16* Wkh  = Wqh + 65536;
    _Float16* Wvh  = Wkh + 65536;
    _Float16* Kh   = Wvh + 65536;                             // 1 MB
    _Float16* Vt   = Kh + 2048 * 256;                         // 1 MB (transposed V)
    _Float16* Qh   = Vt + 2048 * 256;                         // 33.5 MB

    cvt_w_kernel<<<256, 256, 0, stream>>>(WQ, WK, WV, Wqh, Wkh, Wvh);
    temporal_pool_kernel<<<2048, 256, 0, stream>>>(x, Xh, TAtt);
    // Q = x @ W_Q^T   (M = B*T*N = 65536), row-major out
    gemm_xwT_kernel<<<4096, 128, 0, stream>>>(Xh, Wqh, Qh, 0);
    // K = TAtt @ W_K^T (M = B*N = 2048), row-major out
    gemm_xwT_kernel<<<128, 128, 0, stream>>>(TAtt, Wkh, Kh, 0);
    // V = TAtt @ W_V^T, stored transposed per batch: Vt[b][h*32+d][n]
    gemm_xwT_kernel<<<128, 128, 0, stream>>>(TAtt, Wvh, Vt, 1);
    attn_kernel<<<8192, 128, 0, stream>>>(Qh, Kh, Vt, out);
}
